// MultiScaleDeformableAttention_69363721830593
// MI455X (gfx1250) — matrix operand order repaired and checked by reading, not compiled, not run on hardware
//
#include <hip/hip_runtime.h>

typedef __attribute__((ext_vector_type(16))) _Float16 v16h;
typedef __attribute__((ext_vector_type(8)))  float    v8f;
typedef __attribute__((ext_vector_type(4)))  float    v4f;

#define D_MODEL   256
#define N_HEADS   8
#define N_LEVELS  4
#define N_POINTS  4
#define HEAD_DIM  32
#define BATCH     8
#define LQ        2048
#define LV        21760

// ---------------------------------------------------------------------------
// Pre-swizzle a weight matrix W[K x N] (f32, row-major) into f16 fragments in
// a contiguous-k B-operand layout for V_WMMA_F32_16X16X32_F16 (A uses the same k order):
//   frag[kblk][nt][lane][j]  (j = 0..15 contiguous f16, 32 B per lane)
// where element j of lane holds W[kblk*32 + 16*(lane>>4) + j][nt*16 + (lane&15)].
// grid = (K/32, N/16), block = 32 (one lane each). Runs once; cost is trivial.
// ---------------------------------------------------------------------------
__global__ void msda_wswz(const float* __restrict__ W, _Float16* __restrict__ Wf,
                          int N) {
    const int kblk = blockIdx.x, nt = blockIdx.y, lane = threadIdx.x;
    const int half = lane >> 4, nr = lane & 15;
    const float* src = W + (size_t)(kblk * 32 + half * 16) * N + nt * 16 + nr;
    _Float16* dst = Wf + (((size_t)kblk * gridDim.y + nt) * 32 + lane) * 16;
    v16h frag;
#pragma unroll
    for (int j = 0; j < 16; ++j) frag[j] = (_Float16)src[(size_t)j * N];
    *(v16h*)dst = frag;
}

// ---------------------------------------------------------------------------
// C[M x N] = A[M x K] * W[K x N] + bias[N], A/C f32 row-major, W pre-swizzled
// f16 fragments. K == 256 at every call site (8 k-blocks of 32).
//
// Per block (256 thr = 8 waves): the 32 KB B slab (8 kblks x 4 ntiles x 32
// lanes x 32 B) is staged global->LDS with GLOBAL_LOAD_ASYNC_TO_LDS_B128
// (ASYNCcnt), then each wave computes a 16(M) x 64(N) strip: A fragment
// loaded b128 + cvt_pk once per k-step and fed to 4 WMMAs whose B operands
// come from LDS (ds_load_b128) at low latency.
// grid = (M/128, N/64). M % 128 == 0, N % 64 == 0, K == 256.
// ---------------------------------------------------------------------------
__global__ void msda_gemm_f16(const float* __restrict__ A,
                              const _Float16* __restrict__ Wf,
                              const float* __restrict__ bias,
                              float* __restrict__ C,
                              int M, int N, int K) {
    __shared__ _Float16 ldsB[16384];                      // 32 KB
    const int tid  = threadIdx.x;
    const int lane = tid & 31;
    const int wave = tid >> 5;
    const int half = lane >> 4;
    const int mr   = lane & 15;
    const int ntb  = blockIdx.y * 4;                      // first of 4 n-tiles
    const int NT   = N >> 4;                              // total n-tiles

    // ---- async-stage the block's B slab into LDS (2048 x 16B chunks) ----
    {
        const char* wbase = (const char*)Wf + (size_t)ntb * 1024;
        const size_t kstride = (size_t)NT * 1024;         // bytes per kblk
        const unsigned ldsbase = (unsigned)(uintptr_t)&ldsB[0];
#pragma unroll
        for (int i = 0; i < 8; ++i) {
            const int e    = i * 256 + tid;               // chunk index
            const int kblk = e >> 8;
            const int r    = e & 255;
            const char* gsrc = wbase + (size_t)kblk * kstride + (size_t)r * 16;
            const unsigned dst = ldsbase + (unsigned)e * 16;
            asm volatile("global_load_async_to_lds_b128 %0, %1, off"
                         :: "v"(dst), "v"(gsrc) : "memory");
        }
        asm volatile("s_wait_asynccnt 0" ::: "memory");
    }
    __syncthreads();

    const int m = blockIdx.x * 128 + wave * 16 + mr;      // A row of this lane
    const float* arow = A + (size_t)m * K;

    v8f acc0 = {}, acc1 = {}, acc2 = {}, acc3 = {};

    for (int k0 = 0; k0 < K; k0 += 32) {
        // A fragment: lane holds K = [k0+16h, k0+16h+16) contiguous
        const v4f* ap0 = (const v4f*)(arow + k0 + half * 16);
        const v4f* ap1 = (const v4f*)(arow + k0 + half * 16 + 8);
        const v4f a0 = ap0[0], a1 = ap0[1];
        const v4f a2 = ap1[0], a3 = ap1[1];
        __builtin_prefetch(arow + k0 + 64, 0, 1);         // stream A ahead
        v16h a;
#pragma unroll
        for (int j = 0; j < 4; ++j) {
            a[j]      = (_Float16)a0[j];
            a[4 + j]  = (_Float16)a1[j];
            a[8 + j]  = (_Float16)a2[j];
            a[12 + j] = (_Float16)a3[j];
        }
        // B fragments from LDS: 32 contiguous bytes per lane per n-tile
        const _Float16* lb = &ldsB[(size_t)(k0 >> 5) * 2048 + lane * 16];
        const v16h b0 = *(const v16h*)(lb);
        const v16h b1 = *(const v16h*)(lb + 512);
        const v16h b2 = *(const v16h*)(lb + 1024);
        const v16h b3 = *(const v16h*)(lb + 1536);

        acc0 = __builtin_amdgcn_wmma_f32_16x16x32_f16(false, a, false, b0,
                                                      (short)0, acc0, false, false);
        acc1 = __builtin_amdgcn_wmma_f32_16x16x32_f16(false, a, false, b1,
                                                      (short)0, acc1, false, false);
        acc2 = __builtin_amdgcn_wmma_f32_16x16x32_f16(false, a, false, b2,
                                                      (short)0, acc2, false, false);
        acc3 = __builtin_amdgcn_wmma_f32_16x16x32_f16(false, a, false, b3,
                                                      (short)0, acc3, false, false);
    }

    // epilogue: D layout -> VGPR i holds row (tilebase + i + 8*half), col mr
    const int rowbase = blockIdx.x * 128 + wave * 16 + 8 * half;
    v8f accs[4] = {acc0, acc1, acc2, acc3};
#pragma unroll
    for (int t = 0; t < 4; ++t) {
        const int n  = (ntb + t) * 16 + mr;
        const float bn = bias[n];
#pragma unroll
        for (int i = 0; i < 8; ++i)
            C[(size_t)(rowbase + i) * N + n] = accs[t][i] + bn;
    }
}

// ---------------------------------------------------------------------------
// Sampling core: one wave32 per (b, q, head); lane = channel (HEAD_DIM == 32).
// Per-wave uniform state (logits, offsets, ref points) is loaded through
// readfirstlane-forced scalar addresses -> SMEM loads; vector-memory slots
// stay free for the bilinear gathers (coalesced 128 B per sample point).
// grid_sample(align_corners=False) semantics: pixel = loc*size - 0.5, zero pad.
// ---------------------------------------------------------------------------
__global__ void msda_sample(const float* __restrict__ vproj,   // (B, LV, 256)
                            const float* __restrict__ refp,    // (B, LQ, 4, 2)
                            const float* __restrict__ offs,    // (B*LQ, 256)
                            const float* __restrict__ logits,  // (B*LQ, 128)
                            float* __restrict__ head)          // (B*LQ, 256)
{
    const int wid0 = (blockIdx.x * blockDim.x + threadIdx.x) >> 5;
    const int wid  = __builtin_amdgcn_readfirstlane(wid0);   // wave-uniform
    const int lane = threadIdx.x & 31;

    const int h = wid & 7;
    const int q = (wid >> 3) & (LQ - 1);
    const int b = wid >> 14;                 // 8 heads * 2048 queries = 2^14
    const int row = b * LQ + q;

    const int Hs[4] = {128, 64, 32, 16};
    const int Ss[4] = {0, 16384, 20480, 21504};

    // --- softmax over the 16 (level, point) logits of this head ---
    const float* lg = logits + (size_t)row * (N_HEADS * 16) + h * 16;
    float wts[16];
    float mx = -1e30f;
#pragma unroll
    for (int i = 0; i < 16; ++i) { wts[i] = lg[i]; mx = fmaxf(mx, wts[i]); }
    float sum = 0.f;
#pragma unroll
    for (int i = 0; i < 16; ++i) { wts[i] = __expf(wts[i] - mx); sum += wts[i]; }
    const float inv = 1.f / sum;

    const float* of = offs + (size_t)row * 256 + h * 32;  // [l*8 + p*2 + c]
    const float* rp = refp + (size_t)row * 8;             // [l*2 + c]

    float acc = 0.f;
#pragma unroll
    for (int l = 0; l < N_LEVELS; ++l) {
        const int Hl = Hs[l], Wl = Hs[l];
        const float* base =
            vproj + ((size_t)b * LV + Ss[l]) * D_MODEL + h * HEAD_DIM + lane;
        const float rx = rp[l * 2 + 0];
        const float ry = rp[l * 2 + 1];
#pragma unroll
        for (int p = 0; p < N_POINTS; ++p) {
            const float w  = wts[l * 4 + p] * inv;
            // offsets normalized by H of level 0 (== 128) in the reference
            const float lx = rx + of[l * 8 + p * 2 + 0] * (1.f / 128.f);
            const float ly = ry + of[l * 8 + p * 2 + 1] * (1.f / 128.f);
            const float x  = lx * (float)Wl - 0.5f;
            const float y  = ly * (float)Hl - 0.5f;
            const float xf = floorf(x), yf = floorf(y);
            const float wx = x - xf,    wy = y - yf;
            const int x0 = (int)xf, y0 = (int)yf;

            float v00 = 0.f, v10 = 0.f, v01 = 0.f, v11 = 0.f;
            if (x0 >= 0     && x0 < Wl     && y0 >= 0     && y0 < Hl)
                v00 = base[(size_t)(y0 * Wl + x0) * D_MODEL];
            if (x0 + 1 >= 0 && x0 + 1 < Wl && y0 >= 0     && y0 < Hl)
                v10 = base[(size_t)(y0 * Wl + x0 + 1) * D_MODEL];
            if (x0 >= 0     && x0 < Wl     && y0 + 1 >= 0 && y0 + 1 < Hl)
                v01 = base[(size_t)((y0 + 1) * Wl + x0) * D_MODEL];
            if (x0 + 1 >= 0 && x0 + 1 < Wl && y0 + 1 >= 0 && y0 + 1 < Hl)
                v11 = base[(size_t)((y0 + 1) * Wl + x0 + 1) * D_MODEL];

            const float top = v00 * (1.f - wx) + v10 * wx;
            const float bot = v01 * (1.f - wx) + v11 * wx;
            acc += w * (top * (1.f - wy) + bot * wy);
        }
    }

    head[(size_t)row * D_MODEL + h * HEAD_DIM + lane] = acc;
}

// ---------------------------------------------------------------------------
// Launch: 4 weight pre-swizzles + 3 input GEMMs -> sampling -> output GEMM.
// ---------------------------------------------------------------------------
extern "C" void kernel_launch(void* const* d_in, const int* in_sizes, int n_in,
                              void* d_out, int out_size, void* d_ws, size_t ws_size,
                              hipStream_t stream) {
    (void)in_sizes; (void)n_in; (void)out_size; (void)ws_size;

    const float* query  = (const float*)d_in[0];
    const float* refp   = (const float*)d_in[1];
    const float* value  = (const float*)d_in[2];
    // d_in[3] value_spatial_shapes / d_in[4] value_level_start_index: baked in.
    const float* W_off  = (const float*)d_in[5];
    const float* b_off  = (const float*)d_in[6];
    const float* W_attn = (const float*)d_in[7];
    const float* b_attn = (const float*)d_in[8];
    const float* W_val  = (const float*)d_in[9];
    const float* b_val  = (const float*)d_in[10];
    const float* W_out  = (const float*)d_in[11];
    const float* b_out  = (const float*)d_in[12];
    float* out = (float*)d_out;

    // workspace layout
    float* ws    = (float*)d_ws;
    float* vproj = ws;                                        // 44,564,480 f32
    float* offsb = vproj + (size_t)BATCH * LV * D_MODEL;      //  4,194,304 f32
    float* attn  = offsb + (size_t)BATCH * LQ * 256;          //  2,097,152 f32
    float* headw = attn  + (size_t)BATCH * LQ * 128;          //  4,194,304 f32
    _Float16* wf_val  = (_Float16*)(headw + (size_t)BATCH * LQ * D_MODEL);
    _Float16* wf_off  = wf_val  + 256 * 256;                  // 64 K f16 each
    _Float16* wf_attn = wf_off  + 256 * 256;
    _Float16* wf_out  = wf_attn + 256 * 128;

    const dim3 blk(256);

    // pre-swizzle weights into f16 WMMA B-fragments (tiny, L2-resident after)
    msda_wswz<<<dim3(8, 16), dim3(32), 0, stream>>>(W_val,  wf_val,  256);
    msda_wswz<<<dim3(8, 16), dim3(32), 0, stream>>>(W_off,  wf_off,  256);
    msda_wswz<<<dim3(8,  8), dim3(32), 0, stream>>>(W_attn, wf_attn, 128);
    msda_wswz<<<dim3(8, 16), dim3(32), 0, stream>>>(W_out,  wf_out,  256);

    // vproj = value @ W_val + b_val           M=174080, N=256, K=256
    msda_gemm_f16<<<dim3(174080 / 128, 4), blk, 0, stream>>>(
        value, wf_val, b_val, vproj, BATCH * LV, 256, 256);

    // off = query @ W_off + b_off             M=16384, N=256, K=256
    msda_gemm_f16<<<dim3(16384 / 128, 4), blk, 0, stream>>>(
        query, wf_off, b_off, offsb, BATCH * LQ, 256, 256);

    // attn logits = query @ W_attn + b_attn   M=16384, N=128, K=256
    msda_gemm_f16<<<dim3(16384 / 128, 2), blk, 0, stream>>>(
        query, wf_attn, b_attn, attn, BATCH * LQ, 128, 256);

    // softmax + bilinear sampling + weighted sum: 131072 waves, 8 waves/block
    msda_sample<<<dim3((BATCH * LQ * N_HEADS * 32) / 256), blk, 0, stream>>>(
        vproj, refp, offsb, attn, headw);

    // out = head @ W_out + b_out              M=16384, N=256, K=256
    msda_gemm_f16<<<dim3(16384 / 128, 4), blk, 0, stream>>>(
        headw, wf_out, b_out, out, BATCH * LQ, 256, 256);
}
